// dialted_residual_inception_module_up_74972949119499
// MI455X (gfx1250) — compile-verified, matrix operand order checked by reading, not run on hardware
//
#include <hip/hip_runtime.h>
#include <cstdint>
#include <cstddef>

// ---------------------------------------------------------------------------
// Problem constants (match reference)
// ---------------------------------------------------------------------------
static constexpr int KH_IN   = 128;
static constexpr int KH_OUT  = 128;
static constexpr int KD_MOD  = 256;
static constexpr int KN_ST   = 32;
static constexpr int KB      = 8;
static constexpr int KL      = 2048;
static constexpr int KL2     = 4096;      // output length after stride-2 deconv
static constexpr int KCOLS1  = KB * KL;   // 16384 deconv GEMM columns
static constexpr float KLEAKY = 0.3f;
static constexpr float KEPS   = 1e-5f;
static constexpr int KCHUNK  = 256;       // scan chunk length
static constexpr int KNCHUNK = KL2 / KCHUNK; // 16

typedef float    v8f    __attribute__((ext_vector_type(8)));
typedef _Float16 f16x16 __attribute__((ext_vector_type(16)));
typedef _Float16 f16x8  __attribute__((ext_vector_type(8)));
typedef unsigned int u32x4 __attribute__((ext_vector_type(4)));
typedef int      i32x8  __attribute__((ext_vector_type(8)));
typedef int      i32x4  __attribute__((ext_vector_type(4)));

#ifndef __has_builtin
#define __has_builtin(x) 0
#endif
#if defined(__HIP_DEVICE_COMPILE__) && __has_builtin(__builtin_amdgcn_tensor_load_to_lds) && __has_builtin(__builtin_amdgcn_s_wait_tensorcnt)
#define USE_TDM 1
#else
#define USE_TDM 0
#endif

// ---------------------------------------------------------------------------
// WMMA helpers. CDNA5 f16 WMMA: D(16x16,f32) = A(16x32,f16) * B(32x16,f16) + C
// A fragment (16-bit A 16x32 layout): lane m=lane&15, kh=lane>>4; halves j:
//   K = (j>>3)*16 + kh*8 + (j&7)  -> two contiguous 8-half (16B) LDS reads.
// B staged TRANSPOSED in LDS ([n][k]) so its fragment reads identically.
// ---------------------------------------------------------------------------
__device__ __forceinline__ f16x16 ld_frag(const _Float16* p) {
  f16x8 lo = *(const f16x8*)(p);
  f16x8 hi = *(const f16x8*)(p + 16);
  f16x16 r;
#pragma unroll
  for (int i = 0; i < 8; ++i) { r[i] = lo[i]; r[i + 8] = hi[i]; }
  return r;
}

__device__ __forceinline__ v8f wmma_f16(f16x16 a, f16x16 b, v8f c) {
  return __builtin_amdgcn_wmma_f32_16x16x32_f16(false, a, false, b,
                                                (short)0, c, false, false);
}

// ---------------------------------------------------------------------------
// Kernel 1: deconv tap GEMMs.  T[j,t] (128 x 16384) = Wjt^T (128x128) @ X.
// Grid: x = 128 column-blocks of 128 cols, y = 8 M-tiles of 16, z = 4 branches.
// Block 256 = 8 waves; wave w owns 16 columns; loops both taps (8 WMMA/wave).
// W panel staged via TDM (data_size=8B covers both interleaved taps).
// ---------------------------------------------------------------------------
__global__ __launch_bounds__(256) void deconv_wmma_kernel(
    const float* __restrict__ x,
    const float* __restrict__ w1, const float* __restrict__ w2,
    const float* __restrict__ w3, const float* __restrict__ w4,
    float* __restrict__ T)
{
  __shared__ __align__(16) float2   wstage[16 * 128];        // 16KB (t0,t1) pairs
  __shared__ __align__(16) _Float16 Apan[2][16 * 128];       //  8KB per-tap A panels
  __shared__ __align__(16) _Float16 Xpan[128 * 128];         // 32KB B panel, [col][k]

  const int tid = threadIdx.x;
  const int cb  = blockIdx.x;            // column block
  const int m0  = blockIdx.y * 16;       // Cout tile base
  const int j   = blockIdx.z;            // branch
  const int b   = cb >> 4;               // batch
  const int l0  = (cb & 15) * 128;       // l base within batch

  const float* wj = (j == 0) ? w1 : ((j == 1) ? w2 : ((j == 2) ? w3 : w4));

  // ---- stage W^T panel: wstage[k*16+m] = {w[k][m0+m][0], w[k][m0+m][1]} ----
#if USE_TDM
  if (tid == 0) {
    unsigned ldsa = (unsigned)(uintptr_t)(&wstage[0]);
    unsigned long long ga = (unsigned long long)(uintptr_t)(wj + (size_t)m0 * 2);
    u32x4 g0;
    g0.x = 1u;                                   // count=1, no gather
    g0.y = ldsa;                                 // lds_addr
    g0.z = (unsigned)(ga & 0xffffffffu);         // global_addr[31:0]
    g0.w = (unsigned)((ga >> 32) & 0x1ffffffu) | (2u << 30); // addr[56:32] | type=2
    i32x8 g1;
    const unsigned td0 = 16, td1 = 128;          // tensor dims (8B units)
    g1[0] = (3 << 16);                           // mask=0, data_size=3 (8 bytes)
    g1[1] = (int)((td0 & 0xffffu) << 16);        // tensor_dim0 low16 @bits63:48
    g1[2] = (int)((td0 >> 16) & 0xffffu) | (int)((td1 & 0xffffu) << 16);
    g1[3] = (int)((td1 >> 16) & 0xffffu) | (int)(16u << 16);  // tile_dim0=16
    g1[4] = (int)128;                            // tile_dim1=128, tile_dim2=0
    g1[5] = (int)128;                            // tensor_dim0_stride = 128 (8B units)
    g1[6] = 0;
    g1[7] = 0;                                   // dim1 stride unused (2D tile)
    i32x4 g2 = {0, 0, 0, 0}, g3 = {0, 0, 0, 0};
    i32x8 g4 = {0, 0, 0, 0, 0, 0, 0, 0};         // unused trailing group (6-arg form)
    __builtin_amdgcn_tensor_load_to_lds(g0, g1, g2, g3, g4, 0);
    __builtin_amdgcn_s_wait_tensorcnt(0);
  }
  __syncthreads();
#else
  {
    const float2* wp = (const float2*)wj;        // pairs (tap0,tap1) per (i,o)
    for (int i = tid; i < 16 * 128; i += 256) {
      int m = i & 15, k = i >> 4;
      wstage[i] = wp[(size_t)k * 128 + m0 + m];
    }
    __syncthreads();
  }
#endif

  // convert staged f32 pairs -> per-tap f16 A panels [m][k]
  for (int i = tid; i < 16 * 128; i += 256) {
    int m = i & 15, k = i >> 4;
    float2 v = wstage[i];
    Apan[0][m * 128 + k] = (_Float16)v.x;
    Apan[1][m * 128 + k] = (_Float16)v.y;
  }

  // ---- stage X panel transposed: Xpan[colLocal][k], f16 ----
  for (int i = tid; i < 128 * 32; i += 256) {    // 128 k-rows x 32 float4 groups
    int k  = i >> 5;
    int c4 = (i & 31) * 4;
    float4 vv = *(const float4*)(x + ((size_t)(b * KH_IN + k)) * KL + l0 + c4);
    float tmp[4] = {vv.x, vv.y, vv.z, vv.w};
#pragma unroll
    for (int q = 0; q < 4; ++q)
      Xpan[(c4 + q) * 128 + k] = (_Float16)tmp[q];
  }
  __syncthreads();

  // ---- WMMA GEMM: K=128 in 4 steps of 32, both taps ----
  const int wv   = tid >> 5;
  const int lane = tid & 31;
  const int m    = lane & 15;
  const int kh   = lane >> 4;
  const int n    = lane & 15;
  const int ncol = wv * 16;                      // wave's local column base

#pragma unroll
  for (int t = 0; t < 2; ++t) {
    v8f acc = {};
#pragma unroll
    for (int ks = 0; ks < 4; ++ks) {
      const int kb = ks * 32;
      f16x16 a = ld_frag(&Apan[t][m * 128 + kb + kh * 8]);
      f16x16 bm = ld_frag(&Xpan[(ncol + n) * 128 + kb + kh * 8]);
      acc = wmma_f16(a, bm, acc);
    }
    float* Tout = T + (size_t)(j * 2 + t) * KH_OUT * KCOLS1;
    const int col = cb * 128 + ncol + n;         // == b*2048 + l
#pragma unroll
    for (int r = 0; r < 8; ++r) {
      int row = m0 + r + (kh ? 8 : 0);
      Tout[(size_t)row * KCOLS1 + col] = acc[r];
    }
  }
}

// ---------------------------------------------------------------------------
// Kernel 2: assemble deconv outputs (stride-2 interleave + dilation shift),
// add bias, apply leaky (branches 2-4). Writes yraw (B,512,4096) pre-BN.
// ---------------------------------------------------------------------------
__global__ __launch_bounds__(256) void assemble_kernel(
    const float* __restrict__ T,
    const float* __restrict__ bb1, const float* __restrict__ bb2,
    const float* __restrict__ bb3, const float* __restrict__ bb4,
    float* __restrict__ yraw)
{
  const int row = blockIdx.x >> 4;                   // b*512 + ch
  const int p   = ((blockIdx.x & 15) << 8) + threadIdx.x;
  const int ch  = row & 511;
  const int b   = row >> 9;
  const int j   = ch >> 7;
  const int c   = ch & 127;
  const int d   = 1 << j;
  const float* bias = (j == 0) ? bb1 : ((j == 1) ? bb2 : ((j == 2) ? bb3 : bb4));
  const float* T0 = T + (((size_t)(j * 2 + 0) * KH_OUT + c) * KCOLS1 + (size_t)b * KL);
  const float* T1 = T + (((size_t)(j * 2 + 1) * KH_OUT + c) * KCOLS1 + (size_t)b * KL);

  float v = bias[c];
  if (j == 0) {
    v += (p & 1) ? T1[(p - 1) >> 1] : T0[p >> 1];
  } else {
    if ((p & 1) == 0) {
      v += T0[p >> 1];
      if (p >= d) v += T1[(p - d) >> 1];
    }
    v = (v > 0.f) ? v : KLEAKY * v;                  // leaky before BN
  }
  yraw[(size_t)row * KL2 + p] = v;
}

// ---------------------------------------------------------------------------
// Kernel 3: deterministic per-channel batch stats (sum, sumsq over B x L2).
// One block per channel; tree reduction; no atomics.
// ---------------------------------------------------------------------------
__global__ __launch_bounds__(256) void stats_kernel(
    const float* __restrict__ yraw, float* __restrict__ stats)
{
  __shared__ float ssum[256];
  __shared__ float ssq[256];
  const int ch  = blockIdx.x;
  const int tid = threadIdx.x;
  float a = 0.f, q = 0.f;
  for (int i = tid; i < KB * KL2; i += 256) {
    int b = i >> 12, p = i & (KL2 - 1);
    float v = yraw[((size_t)(b * 512 + ch)) * KL2 + p];
    a += v; q += v * v;
  }
  ssum[tid] = a; ssq[tid] = q;
  __syncthreads();
  for (int s = 128; s > 0; s >>= 1) {
    if (tid < s) { ssum[tid] += ssum[tid + s]; ssq[tid] += ssq[tid + s]; }
    __syncthreads();
  }
  if (tid == 0) { stats[ch] = ssum[0]; stats[512 + ch] = ssq[0]; }
}

// ---------------------------------------------------------------------------
// Kernel 4: fold BN affine into the 1x1 conv weights/bias.
//   scale[ch] = gamma * rsqrt(var+eps); shift[ch] = beta - mean*scale
//   w'[o,ch] = conv_w[o,ch]*scale[ch];  b'[o] = conv_b[o] + sum_ch w*shift
// ---------------------------------------------------------------------------
__global__ __launch_bounds__(512) void fold_kernel(
    const float* __restrict__ stats,
    const float* __restrict__ gamma, const float* __restrict__ beta,
    const float* __restrict__ cw, const float* __restrict__ cb,
    float* __restrict__ wfold, float* __restrict__ bfold)
{
  __shared__ float s_scale[512];
  __shared__ float s_shift[512];
  const int tid = threadIdx.x;
  {
    const float cnt = (float)(KB * KL2);
    float su = stats[tid], sq = stats[512 + tid];
    float mu = su / cnt;
    float var = sq / cnt - mu * mu;
    float sc = gamma[tid & 127] * rsqrtf(var + KEPS);
    s_scale[tid] = sc;
    s_shift[tid] = beta[tid & 127] - mu * sc;
  }
  __syncthreads();
  if (tid < 256) {
    float acc = cb[tid];
    for (int ch = 0; ch < 512; ++ch) {
      float wv = cw[(size_t)tid * 512 + ch];
      wfold[(size_t)tid * 512 + ch] = wv * s_scale[ch];
      acc += wv * s_shift[ch];
    }
    bfold[tid] = acc;
  }
}

// ---------------------------------------------------------------------------
// Kernel 5: 1x1 conv GEMM.  u (256 x 32768) = w' (256x512) @ yraw + b'.
// Grid: x = 256 column blocks of 128, y = 16 M-tiles. K=512 chunked by 128.
// ---------------------------------------------------------------------------
__global__ __launch_bounds__(256) void conv1x1_wmma_kernel(
    const float* __restrict__ yraw,
    const float* __restrict__ wfold, const float* __restrict__ bfold,
    float* __restrict__ u)
{
  __shared__ __align__(16) _Float16 Apan[16 * 128];    //  4KB
  __shared__ __align__(16) _Float16 Bpan[128 * 128];   // 32KB, [col][k]

  const int tid = threadIdx.x;
  const int cb  = blockIdx.x;
  const int m0  = blockIdx.y * 16;
  const int b   = cb >> 5;
  const int p0  = (cb & 31) * 128;

  const int wv   = tid >> 5;
  const int lane = tid & 31;
  const int m    = lane & 15;
  const int kh   = lane >> 4;
  const int n    = lane & 15;
  const int ncol = wv * 16;

  v8f acc = {};
#pragma unroll
  for (int kc = 0; kc < 4; ++kc) {
    __syncthreads();
    // A chunk: [m][kk] halves
    for (int i = tid; i < 16 * 128; i += 256) {
      int mm = i >> 7, kk = i & 127;
      Apan[i] = (_Float16)wfold[(size_t)(m0 + mm) * 512 + kc * 128 + kk];
    }
    // B chunk transposed: [col][k]
    for (int i = tid; i < 128 * 32; i += 256) {
      int k  = i >> 5;
      int c4 = (i & 31) * 4;
      float4 vv = *(const float4*)(
          yraw + ((size_t)(b * 512 + kc * 128 + k)) * KL2 + p0 + c4);
      float tmp[4] = {vv.x, vv.y, vv.z, vv.w};
#pragma unroll
      for (int q = 0; q < 4; ++q)
        Bpan[(c4 + q) * 128 + k] = (_Float16)tmp[q];
    }
    __syncthreads();
#pragma unroll
    for (int ks = 0; ks < 4; ++ks) {
      const int kb = ks * 32;
      f16x16 a  = ld_frag(&Apan[m * 128 + kb + kh * 8]);
      f16x16 bm = ld_frag(&Bpan[(ncol + n) * 128 + kb + kh * 8]);
      acc = wmma_f16(a, bm, acc);
    }
  }
#pragma unroll
  for (int r = 0; r < 8; ++r) {
    int row = m0 + r + (kh ? 8 : 0);
    u[((size_t)(b * KD_MOD + row)) * KL2 + p0 + ncol + n] = acc[r] + bfold[row];
  }
}

// ---------------------------------------------------------------------------
// Kernel 6: S4D discretization. Per (h,n): dA=exp(dt*A), dB=(dA-1)/A,
// P=dA^KCHUNK.  params[idx*6] = {dAr,dAi,dBr,dBi,Pr,Pi}
// ---------------------------------------------------------------------------
__global__ __launch_bounds__(256) void s4_params_kernel(
    const float* __restrict__ A_re, const float* __restrict__ A_im,
    const float* __restrict__ log_dt, float* __restrict__ par)
{
  const int idx = blockIdx.x * 256 + threadIdx.x;    // (h,n)
  if (idx >= KD_MOD * KN_ST) return;
  const int h = idx >> 5;
  const float dt  = __expf(log_dt[h]);
  const float ar  = A_re[idx], ai = A_im[idx];
  const float tr  = dt * ar, ti = dt * ai;
  float s, c;
  __sincosf(ti, &s, &c);
  const float er  = __expf(tr);
  const float dAr = er * c, dAi = er * s;
  const float den = ar * ar + ai * ai;
  const float inv = (den > 0.f) ? (1.f / den) : 0.f;
  const float dBr = ((dAr - 1.f) * ar + dAi * ai) * inv;
  const float dBi = (dAi * ar - (dAr - 1.f) * ai) * inv;
  float ps, pc;
  __sincosf((float)KCHUNK * ti, &ps, &pc);
  const float ep = __expf((float)KCHUNK * tr);
  par[(size_t)idx * 6 + 0] = dAr;
  par[(size_t)idx * 6 + 1] = dAi;
  par[(size_t)idx * 6 + 2] = dBr;
  par[(size_t)idx * 6 + 3] = dBi;
  par[(size_t)idx * 6 + 4] = ep * pc;
  par[(size_t)idx * 6 + 5] = ep * ps;
}

// ---------------------------------------------------------------------------
// Kernel 7: scan pass 1 — per (b,h,n,chunk) local scan from zero state;
// store end-of-chunk state. Wave = 32 states of one (b,h,chunk): u broadcast.
// ---------------------------------------------------------------------------
__global__ __launch_bounds__(256) void scan1_kernel(
    const float* __restrict__ u, const float* __restrict__ par,
    float* __restrict__ car)
{
  const int idx = blockIdx.x * 256 + threadIdx.x;    // ((b*256+h)*16+c)*32+n
  const int n   = idx & 31;
  const int c   = (idx >> 5) & 15;
  const int bh  = idx >> 9;
  const int h   = bh & 255;
  const size_t pb = (size_t)(h * KN_ST + n) * 6;
  const float dAr = par[pb + 0], dAi = par[pb + 1];
  const float dBr = par[pb + 2], dBi = par[pb + 3];
  const float* up = u + (size_t)bh * KL2 + c * KCHUNK;
  float sr = 0.f, si = 0.f;
  for (int t = 0; t < KCHUNK; ++t) {
    if ((t & 63) == 0) __builtin_prefetch(up + t + 64, 0, 0);
    const float uv = up[t];
    const float nr = dAr * sr - dAi * si + dBr * uv;
    const float ni = dAr * si + dAi * sr + dBi * uv;
    sr = nr; si = ni;
  }
  const size_t ci = ((size_t)(bh * KN_ST + n) * KNCHUNK + c) * 2;
  car[ci + 0] = sr;
  car[ci + 1] = si;
}

// ---------------------------------------------------------------------------
// Kernel 8: carry fix-up — sequential over the 16 chunks per (b,h,n):
// replace local end-states with corrected chunk INITIAL states.
// ---------------------------------------------------------------------------
__global__ __launch_bounds__(256) void scan_fix_kernel(
    const float* __restrict__ par, float* __restrict__ car)
{
  const int idx = blockIdx.x * 256 + threadIdx.x;    // bh*32 + n
  const int n   = idx & 31;
  const int h   = (idx >> 5) & 255;
  const size_t pb = (size_t)(h * KN_ST + n) * 6;
  const float Pr = par[pb + 4], Pi = par[pb + 5];
  const size_t base = (size_t)idx * KNCHUNK * 2;
  float sr = 0.f, si = 0.f;
  for (int c = 0; c < KNCHUNK; ++c) {
    const float tr = car[base + c * 2 + 0];
    const float ti = car[base + c * 2 + 1];
    car[base + c * 2 + 0] = sr;                      // corrected initial state
    car[base + c * 2 + 1] = si;
    const float nr = Pr * sr - Pi * si + tr;
    const float ni = Pr * si + Pi * sr + ti;
    sr = nr; si = ni;
  }
}

// ---------------------------------------------------------------------------
// Kernel 9: scan pass 2 — re-scan each chunk from its corrected initial state
// and emit y = 2*Re(C·s) + D*u.  Wave = (b,h,chunk), lane = state n.
// Per-step cross-lane sum deferred through LDS in 32-step tiles.
// ---------------------------------------------------------------------------
__global__ __launch_bounds__(256) void scan2_kernel(
    const float* __restrict__ u, const float* __restrict__ par,
    const float* __restrict__ car,
    const float* __restrict__ C_re, const float* __restrict__ C_im,
    const float* __restrict__ Dv, float* __restrict__ out)
{
  __shared__ float red[8][32 * 32];                  // 32KB, per-wave tiles
  const int tid  = threadIdx.x;
  const int wv   = tid >> 5;
  const int lane = tid & 31;
  const int g    = blockIdx.x * 8 + wv;              // (b*256+h)*16 + c
  const int c    = g & 15;
  const int bh   = g >> 4;
  const int h    = bh & 255;
  const int n    = lane;
  const size_t pb = (size_t)(h * KN_ST + n) * 6;
  const float dAr = par[pb + 0], dAi = par[pb + 1];
  const float dBr = par[pb + 2], dBi = par[pb + 3];
  const float Cr  = C_re[h * KN_ST + n];
  const float Ci  = C_im[h * KN_ST + n];
  const float Dh  = Dv[h];
  const size_t ci = ((size_t)(bh * KN_ST + n) * KNCHUNK + c) * 2;
  float sr = car[ci + 0], si = car[ci + 1];
  const size_t ub = (size_t)bh * KL2 + (size_t)c * KCHUNK;

  for (int tb = 0; tb < KCHUNK / 32; ++tb) {
#pragma unroll 8
    for (int tt = 0; tt < 32; ++tt) {
      const float uv = u[ub + tb * 32 + tt];
      const float nr = dAr * sr - dAi * si + dBr * uv;
      const float ni = dAr * si + dAi * sr + dBi * uv;
      sr = nr; si = ni;
      red[wv][tt * 32 + n] = Cr * sr - Ci * si;      // Re(C·s) contribution
    }
    __syncthreads();
    float acc = 0.f;
#pragma unroll
    for (int n2 = 0; n2 < 32; ++n2) acc += red[wv][lane * 32 + n2];
    const float uvl = u[ub + tb * 32 + lane];
    out[ub + tb * 32 + lane] = 2.f * acc + Dh * uvl;
    __syncthreads();
  }
}

// ---------------------------------------------------------------------------
// Launch. Workspace layout (floats):
//   T      : 8*128*16384   = 16,777,216   (64 MB)
//   yraw   : 8*512*4096    = 16,777,216   (64 MB)
//   stats  : 1024
//   wfold  : 256*512       = 131,072
//   bfold  : 256
//   u      : 8*256*4096    = 8,388,608    (32 MB)
//   params : 256*32*6      = 49,152
//   carries: 8*256*32*16*2 = 2,097,152    ( 8 MB)   -> ~176 MB total
// ---------------------------------------------------------------------------
extern "C" void kernel_launch(void* const* d_in, const int* in_sizes, int n_in,
                              void* d_out, int out_size, void* d_ws, size_t ws_size,
                              hipStream_t stream) {
  (void)in_sizes; (void)n_in; (void)out_size; (void)ws_size;
  const float* x      = (const float*)d_in[0];
  const float* w1     = (const float*)d_in[1];
  const float* b1     = (const float*)d_in[2];
  const float* w2     = (const float*)d_in[3];
  const float* b2     = (const float*)d_in[4];
  const float* w3     = (const float*)d_in[5];
  const float* b3     = (const float*)d_in[6];
  const float* w4     = (const float*)d_in[7];
  const float* b4     = (const float*)d_in[8];
  const float* gamma  = (const float*)d_in[9];
  const float* beta   = (const float*)d_in[10];
  const float* conv_w = (const float*)d_in[11];
  const float* conv_b = (const float*)d_in[12];
  const float* A_re   = (const float*)d_in[13];
  const float* A_im   = (const float*)d_in[14];
  const float* log_dt = (const float*)d_in[15];
  const float* C_re   = (const float*)d_in[16];
  const float* C_im   = (const float*)d_in[17];
  const float* Dv     = (const float*)d_in[18];

  float* ws    = (float*)d_ws;
  float* T     = ws;
  float* yraw  = T + (size_t)16777216;
  float* stats = yraw + (size_t)16777216;
  float* wfold = stats + 1024;
  float* bfold = wfold + 131072;
  float* u     = bfold + 256;
  float* par   = u + (size_t)8388608;
  float* car   = par + 49152;

  deconv_wmma_kernel<<<dim3(128, 8, 4), 256, 0, stream>>>(x, w1, w2, w3, w4, T);
  assemble_kernel<<<dim3(65536), 256, 0, stream>>>(T, b1, b2, b3, b4, yraw);
  stats_kernel<<<dim3(512), 256, 0, stream>>>(yraw, stats);
  fold_kernel<<<dim3(1), 512, 0, stream>>>(stats, gamma, beta, conv_w, conv_b,
                                           wfold, bfold);
  conv1x1_wmma_kernel<<<dim3(256, 16), 256, 0, stream>>>(yraw, wfold, bfold, u);
  s4_params_kernel<<<dim3(32), 256, 0, stream>>>(A_re, A_im, log_dt, par);
  scan1_kernel<<<dim3(4096), 256, 0, stream>>>(u, par, car);
  scan_fix_kernel<<<dim3(256), 256, 0, stream>>>(par, car);
  scan2_kernel<<<dim3(4096), 256, 0, stream>>>(u, par, car, C_re, C_im, Dv,
                                               (float*)d_out);
}
